// AutoregressiveModel_6528350290154
// MI455X (gfx1250) — matrix operand order repaired and checked by reading, not compiled, not run on hardware
//
#include <hip/hip_runtime.h>
#include <hip/hip_bf16.h>

// ---------------------------------------------------------------------------
// Transformer encoder layer + vocab head for MI455X (gfx1250, wave32, WMMA).
// All heavy GEMMs run on v_wmma_f32_16x16x32_bf16 with fp32 accumulation.
// GEMM: 32x64 wave tile, software-pipelined K loop with peeled final step
// (no conditional in the steady state -> no dead zero-init VALU traffic).
// ---------------------------------------------------------------------------

#define HW_DIM 32
#define SEQ    1024
#define WIN    9
#define DMODEL 512
#define DFF    1024
#define VOCAB  16384
#define BATCH  8
#define MTOT   (BATCH * SEQ)   // 8192 rows

typedef __attribute__((ext_vector_type(16))) __bf16 v16bf;
typedef __attribute__((ext_vector_type(8)))  __bf16 v8bf;
typedef __attribute__((ext_vector_type(8)))  float  v8f;

static __device__ __forceinline__ int iabs_(int v) { return v < 0 ? -v : v; }

// A fragment (16x32 bf16): lane l -> row (l&15), K-range
// (l>>4)*16 + [0..15]: two contiguous 16B loads.
static __device__ __forceinline__ v16bf load_a(const __bf16* p) {
    v8bf lo = *(const v8bf*)p;
    v8bf hi = *(const v8bf*)(p + 8);
    v16bf t;
#pragma unroll
    for (int i = 0; i < 8; ++i) { t[i] = lo[i]; t[i + 8] = hi[i]; }
    return t;
}
// B fragment (32x16 bf16): lane l -> col (l&15), K-range (l>>4)*16 + [0..15]:
// one contiguous 32B load from the K-major (transposed) operand.
static __device__ __forceinline__ v16bf load_b(const __bf16* p) {
    return *(const v16bf*)p;
}
static __device__ __forceinline__ v8f wmma_bf16(v16bf a, v16bf b, v8f c) {
    return __builtin_amdgcn_wmma_f32_16x16x32_bf16(false, a, false, b,
                                                   (short)0, c, false, false);
}

// ---------------------------------------------------------------------------
// Weight convert + transpose:  W[K,N] fp32 (row-major)  ->  Wt[N,K] bf16
// ---------------------------------------------------------------------------
__global__ void k_transpose_w(const float* __restrict__ W,
                              __bf16* __restrict__ Wt, int K, int N) {
    size_t i = (size_t)blockIdx.x * blockDim.x + threadIdx.x;
    size_t total = (size_t)K * N;
    if (i >= total) return;
    int n = (int)(i % N);
    int k = (int)(i / N);
    Wt[(size_t)n * K + k] = (__bf16)W[i];
}

// ---------------------------------------------------------------------------
// Embedding + 2D positional encoding -> h0 (fp32 for residual, bf16 for GEMM)
// ---------------------------------------------------------------------------
__global__ void k_embed(const int* __restrict__ x, const float* __restrict__ tab,
                        const float* __restrict__ rowe, const float* __restrict__ cole,
                        float* __restrict__ h0f, __bf16* __restrict__ h0b) {
    size_t i = (size_t)blockIdx.x * blockDim.x + threadIdx.x;
    if (i >= (size_t)MTOT * DMODEL) return;
    int d = (int)(i % DMODEL);
    size_t t = i / DMODEL;
    int s = (int)(t % SEQ);
    int r = s >> 5, c = s & 31;
    float pos = (d < DMODEL / 2) ? rowe[r * (DMODEL / 2) + d]
                                 : cole[c * (DMODEL / 2) + (d - DMODEL / 2)];
    int tok = x[t];
    float v = tab[(size_t)tok * DMODEL + d] + pos;
    h0f[i] = v;
    h0b[i] = (__bf16)v;
}

// ---------------------------------------------------------------------------
// Generic batched WMMA GEMM:
//   C[m,n] = alpha * sum_k A[m,k] * Bt[n,k] + bias[n]   (opt ReLU, opt NT store)
// A  : bf16 [Mdim,Kdim] row-major (+ batch stride sA)
// Bt : bf16 [Ndim,Kdim] row-major (+ batch stride sB)  == B transposed
// Out: fp32 Cf and/or bf16 Cb, [Mdim,Ndim] (+ batch stride sC)
//
// Block = 128 threads = 4 waves; block tile 128x64; wave tile 32x64:
// 2 A-fragments x 4 B-fragments -> 8 independent WMMAs per K=32 step.
// Next-step fragments are loaded unconditionally before the current WMMAs;
// the final K step is peeled so the steady-state loop has no branches and
// no dead zero-init moves.
// ---------------------------------------------------------------------------
__global__ __launch_bounds__(128)
void k_gemm(const __bf16* __restrict__ A, long long sA,
            const __bf16* __restrict__ Bt, long long sB,
            const float* __restrict__ bias,
            float alpha, int relu, int ntstore,
            float* __restrict__ Cf, __bf16* __restrict__ Cb, long long sC,
            int Mdim, int Ndim, int Kdim) {
    const int wave  = threadIdx.x >> 5;
    const int lane  = threadIdx.x & 31;
    const int half  = lane >> 4;   // 0 or 1
    const int l15   = lane & 15;
    const int batch = blockIdx.z;

    A  += (size_t)batch * sA;
    Bt += (size_t)batch * sB;

    const int m0 = blockIdx.y * 128 + wave * 32;
    const int n0 = blockIdx.x * 64;

    const __bf16* pa0 = A + (size_t)(m0 + l15) * Kdim + half * 16;
    const __bf16* pa1 = pa0 + (size_t)16 * Kdim;
    const __bf16* pb0 = Bt + (size_t)(n0 + l15) * Kdim + half * 16;
    const __bf16* pb1 = pb0 + (size_t)16 * Kdim;
    const __bf16* pb2 = pb0 + (size_t)32 * Kdim;
    const __bf16* pb3 = pb0 + (size_t)48 * Kdim;

    v8f acc00 = {}, acc01 = {}, acc02 = {}, acc03 = {};
    v8f acc10 = {}, acc11 = {}, acc12 = {}, acc13 = {};

    // prologue: K-step 0 fragments
    v16bf a0 = load_a(pa0), a1 = load_a(pa1);
    v16bf b0 = load_b(pb0), b1 = load_b(pb1);
    v16bf b2 = load_b(pb2), b3 = load_b(pb3);

    int k0 = 0;
    for (; k0 + 32 < Kdim; k0 += 32) {
        const int kn = k0 + 32;
        // next-step fragment loads: no dependence on current WMMAs
        v16bf a0n = load_a(pa0 + kn);
        v16bf a1n = load_a(pa1 + kn);
        v16bf b0n = load_b(pb0 + kn);
        v16bf b1n = load_b(pb1 + kn);
        v16bf b2n = load_b(pb2 + kn);
        v16bf b3n = load_b(pb3 + kn);
        // speculative gfx1250 global_prefetch_b8 one strip further ahead
        __builtin_prefetch(pb0 + kn + 32, 0, 3);
        __builtin_prefetch(pb1 + kn + 32, 0, 3);
        __builtin_prefetch(pb2 + kn + 32, 0, 3);
        __builtin_prefetch(pb3 + kn + 32, 0, 3);

        acc00 = wmma_bf16(a0, b0, acc00);
        acc01 = wmma_bf16(a0, b1, acc01);
        acc02 = wmma_bf16(a0, b2, acc02);
        acc03 = wmma_bf16(a0, b3, acc03);
        acc10 = wmma_bf16(a1, b0, acc10);
        acc11 = wmma_bf16(a1, b1, acc11);
        acc12 = wmma_bf16(a1, b2, acc12);
        acc13 = wmma_bf16(a1, b3, acc13);

        a0 = a0n; a1 = a1n;
        b0 = b0n; b1 = b1n; b2 = b2n; b3 = b3n;
    }
    // peeled final K step (fragments already resident)
    acc00 = wmma_bf16(a0, b0, acc00);
    acc01 = wmma_bf16(a0, b1, acc01);
    acc02 = wmma_bf16(a0, b2, acc02);
    acc03 = wmma_bf16(a0, b3, acc03);
    acc10 = wmma_bf16(a1, b0, acc10);
    acc11 = wmma_bf16(a1, b1, acc11);
    acc12 = wmma_bf16(a1, b2, acc12);
    acc13 = wmma_bf16(a1, b3, acc13);

    float*  cf = Cf ? Cf + (size_t)batch * sC : nullptr;
    __bf16* cb = Cb ? Cb + (size_t)batch * sC : nullptr;

    v8f acc[2][4] = {{acc00, acc01, acc02, acc03},
                     {acc10, acc11, acc12, acc13}};
#pragma unroll
    for (int rblk = 0; rblk < 2; ++rblk) {
#pragma unroll
        for (int t = 0; t < 4; ++t) {
            int col = n0 + 16 * t + l15;
            float bterm = bias ? bias[col] : 0.0f;
#pragma unroll
            for (int i = 0; i < 8; ++i) {
                int row = m0 + 16 * rblk + half * 8 + i;
                float v = acc[rblk][t][i] * alpha + bterm;
                if (relu) v = v > 0.0f ? v : 0.0f;
                size_t idx = (size_t)row * Ndim + col;
                if (cf) {
                    if (ntstore) __builtin_nontemporal_store(v, &cf[idx]);
                    else         cf[idx] = v;
                }
                if (cb) cb[idx] = (__bf16)v;
            }
        }
    }
}

// ---------------------------------------------------------------------------
// Batched bf16 activation transpose: in [B,rows,cols] -> out [B,cols,rows]
// ---------------------------------------------------------------------------
__global__ void k_transpose_act(const __bf16* __restrict__ in,
                                __bf16* __restrict__ out, int rows, int cols) {
    __shared__ unsigned short tile[32][33];
    int b = blockIdx.z;
    const unsigned short* inp = (const unsigned short*)in + (size_t)b * rows * cols;
    unsigned short* outp = (unsigned short*)out + (size_t)b * rows * cols;
    int c0 = blockIdx.x * 32;
    int r0 = blockIdx.y * 32;
    int tx = threadIdx.x, ty = threadIdx.y;   // 32 x 8
#pragma unroll
    for (int i = 0; i < 32; i += 8)
        tile[ty + i][tx] = inp[(size_t)(r0 + ty + i) * cols + c0 + tx];
    __syncthreads();
#pragma unroll
    for (int i = 0; i < 32; i += 8)
        outp[(size_t)(c0 + ty + i) * rows + r0 + tx] = tile[tx][ty + i];
}

// ---------------------------------------------------------------------------
// Masked softmax over score rows; local-causal window mask computed
// analytically. One block (256 threads = 8 waves) per row of SEQ=1024.
// ---------------------------------------------------------------------------
__global__ __launch_bounds__(256)
void k_softmax(const float* __restrict__ scores, __bf16* __restrict__ attn) {
    const int row = blockIdx.x;          // 0 .. B*SEQ-1
    const int s = row % SEQ;
    const int r = s >> 5, c = s & 31;
    const float* src = scores + (size_t)row * SEQ;
    __bf16* dst = attn + (size_t)row * SEQ;

    __shared__ float red[256];
    const int tid = threadIdx.x;
    const float NEG = -3.0e38f;

    float vals[4];
    float mx = NEG;
#pragma unroll
    for (int i = 0; i < 4; ++i) {
        int j = tid + i * 256;
        int rj = j >> 5, cj = j & 31;
        bool allow = (iabs_(r - rj) <= WIN) && (iabs_(c - cj) <= WIN) && (j <= s);
        float v = allow ? src[j] : NEG;
        vals[i] = v;
        mx = fmaxf(mx, v);
    }
    red[tid] = mx; __syncthreads();
    for (int off = 128; off > 0; off >>= 1) {
        if (tid < off) red[tid] = fmaxf(red[tid], red[tid + off]);
        __syncthreads();
    }
    mx = red[0]; __syncthreads();

    float sum = 0.0f;
#pragma unroll
    for (int i = 0; i < 4; ++i) {
        float e = __expf(vals[i] - mx);
        vals[i] = e;
        sum += e;
    }
    red[tid] = sum; __syncthreads();
    for (int off = 128; off > 0; off >>= 1) {
        if (tid < off) red[tid] += red[tid + off];
        __syncthreads();
    }
    float inv = 1.0f / red[0];
#pragma unroll
    for (int i = 0; i < 4; ++i)
        dst[tid + i * 256] = (__bf16)(vals[i] * inv);
}

// ---------------------------------------------------------------------------
// out = LayerNorm(x + y) * g + b; fp32 out (nullable) + bf16 out (nullable).
// One block (128 threads) per row of D=512.
// ---------------------------------------------------------------------------
__global__ __launch_bounds__(128)
void k_add_ln(const float* __restrict__ x, const float* __restrict__ y,
              const float* __restrict__ g, const float* __restrict__ b,
              float* __restrict__ outf, __bf16* __restrict__ outb) {
    const int row = blockIdx.x;
    const float* xr = x + (size_t)row * DMODEL;
    const float* yr = y + (size_t)row * DMODEL;
    const int tid = threadIdx.x;

    __shared__ float red[128];
    float v[4];
    float s = 0.0f;
#pragma unroll
    for (int i = 0; i < 4; ++i) {
        int j = tid + i * 128;
        v[i] = xr[j] + yr[j];
        s += v[i];
    }
    red[tid] = s; __syncthreads();
    for (int off = 64; off > 0; off >>= 1) {
        if (tid < off) red[tid] += red[tid + off];
        __syncthreads();
    }
    float mu = red[0] * (1.0f / DMODEL); __syncthreads();

    float ss = 0.0f;
#pragma unroll
    for (int i = 0; i < 4; ++i) { float d = v[i] - mu; ss += d * d; }
    red[tid] = ss; __syncthreads();
    for (int off = 64; off > 0; off >>= 1) {
        if (tid < off) red[tid] += red[tid + off];
        __syncthreads();
    }
    float rstd = rsqrtf(red[0] * (1.0f / DMODEL) + 1e-5f);

#pragma unroll
    for (int i = 0; i < 4; ++i) {
        int j = tid + i * 128;
        float o = (v[i] - mu) * rstd * g[j] + b[j];
        if (outf) outf[(size_t)row * DMODEL + j] = o;
        if (outb) outb[(size_t)row * DMODEL + j] = (__bf16)o;
    }
}

// ---------------------------------------------------------------------------
// Host-side orchestration
// ---------------------------------------------------------------------------
extern "C" void kernel_launch(void* const* d_in, const int* in_sizes, int n_in,
                              void* d_out, int out_size, void* d_ws, size_t ws_size,
                              hipStream_t stream) {
    (void)in_sizes; (void)n_in; (void)out_size; (void)ws_size;

    const int*   x    = (const int*)d_in[0];
    const float* tab  = (const float*)d_in[1];
    const float* rowe = (const float*)d_in[2];
    const float* cole = (const float*)d_in[3];
    const float* Wq = (const float*)d_in[4];   const float* bq = (const float*)d_in[5];
    const float* Wk = (const float*)d_in[6];   const float* bk = (const float*)d_in[7];
    const float* Wv = (const float*)d_in[8];   const float* bv = (const float*)d_in[9];
    const float* Wo = (const float*)d_in[10];  const float* bo = (const float*)d_in[11];
    const float* ln1g = (const float*)d_in[12]; const float* ln1b = (const float*)d_in[13];
    const float* W1 = (const float*)d_in[14];  const float* b1 = (const float*)d_in[15];
    const float* W2 = (const float*)d_in[16];  const float* b2 = (const float*)d_in[17];
    const float* ln2g = (const float*)d_in[18]; const float* ln2b = (const float*)d_in[19];
    const float* Wh = (const float*)d_in[20];  const float* bh = (const float*)d_in[21];
    float* out = (float*)d_out;

    // ---- workspace layout (~212 MB, 256-byte aligned slices) ----
    char* ws = (char*)d_ws;
    size_t off = 0;
    auto alloc = [&](size_t bytes) -> void* {
        void* p = ws + off;
        off = (off + bytes + 255) & ~(size_t)255;
        return p;
    };
    float*  h0f = (float*) alloc((size_t)MTOT * DMODEL * 4);
    __bf16* h0b = (__bf16*)alloc((size_t)MTOT * DMODEL * 2);
    __bf16* WqT = (__bf16*)alloc((size_t)DMODEL * DMODEL * 2);
    __bf16* WkT = (__bf16*)alloc((size_t)DMODEL * DMODEL * 2);
    __bf16* WvT = (__bf16*)alloc((size_t)DMODEL * DMODEL * 2);
    __bf16* WoT = (__bf16*)alloc((size_t)DMODEL * DMODEL * 2);
    __bf16* W1T = (__bf16*)alloc((size_t)DFF * DMODEL * 2);
    __bf16* W2T = (__bf16*)alloc((size_t)DMODEL * DFF * 2);
    __bf16* WhT = (__bf16*)alloc((size_t)VOCAB * DMODEL * 2);
    __bf16* qb  = (__bf16*)alloc((size_t)MTOT * DMODEL * 2);
    __bf16* kb  = (__bf16*)alloc((size_t)MTOT * DMODEL * 2);
    __bf16* vb  = (__bf16*)alloc((size_t)MTOT * DMODEL * 2);
    __bf16* vT  = (__bf16*)alloc((size_t)MTOT * DMODEL * 2);
    float*  sc  = (float*) alloc((size_t)BATCH * SEQ * SEQ * 4);
    __bf16* at  = (__bf16*)alloc((size_t)BATCH * SEQ * SEQ * 2);
    __bf16* oin = (__bf16*)alloc((size_t)MTOT * DMODEL * 2);
    float*  of  = (float*) alloc((size_t)MTOT * DMODEL * 4);
    float*  h1f = (float*) alloc((size_t)MTOT * DMODEL * 4);
    __bf16* h1b = (__bf16*)alloc((size_t)MTOT * DMODEL * 2);
    __bf16* fm  = (__bf16*)alloc((size_t)MTOT * DFF * 2);
    float*  ff  = (float*) alloc((size_t)MTOT * DMODEL * 4);
    __bf16* h2b = (__bf16*)alloc((size_t)MTOT * DMODEL * 2);

    // ---- weight transposes (fp32 -> bf16 [N,K]) ----
    auto tw = [&](const float* W, __bf16* Wt, int K, int N) {
        size_t tot = (size_t)K * N;
        k_transpose_w<<<dim3((unsigned)((tot + 255) / 256)), dim3(256), 0, stream>>>(W, Wt, K, N);
    };
    tw(Wq, WqT, DMODEL, DMODEL);
    tw(Wk, WkT, DMODEL, DMODEL);
    tw(Wv, WvT, DMODEL, DMODEL);
    tw(Wo, WoT, DMODEL, DMODEL);
    tw(W1, W1T, DMODEL, DFF);     // [512,1024] -> [1024,512]
    tw(W2, W2T, DFF, DMODEL);     // [1024,512] -> [512,1024]
    tw(Wh, WhT, DMODEL, VOCAB);   // [512,16384] -> [16384,512]

    // ---- embedding + positional ----
    k_embed<<<dim3((MTOT * DMODEL + 255) / 256), dim3(256), 0, stream>>>(
        x, tab, rowe, cole, h0f, h0b);

    // ---- GEMM launcher: block tile 128x64 ----
    const dim3 gblk(128, 1, 1);
    auto gemm = [&](const __bf16* A, long long sA, const __bf16* Bt, long long sB,
                    const float* bias, float alpha, int relu, int nt,
                    float* Cf, __bf16* Cb, long long sC,
                    int Md, int Nd, int Kd, int nb) {
        dim3 grid(Nd / 64, Md / 128, nb);
        k_gemm<<<grid, gblk, 0, stream>>>(A, sA, Bt, sB, bias, alpha, relu, nt,
                                          Cf, Cb, sC, Md, Nd, Kd);
    };

    // q, k, v projections (bf16 outputs feed attention GEMMs)
    gemm(h0b, 0, WqT, 0, bq, 1.0f, 0, 0, nullptr, qb, 0, MTOT, DMODEL, DMODEL, 1);
    gemm(h0b, 0, WkT, 0, bk, 1.0f, 0, 0, nullptr, kb, 0, MTOT, DMODEL, DMODEL, 1);
    gemm(h0b, 0, WvT, 0, bv, 1.0f, 0, 0, nullptr, vb, 0, MTOT, DMODEL, DMODEL, 1);

    // v^T per batch (for attn@v B-fragments)
    k_transpose_act<<<dim3(DMODEL / 32, SEQ / 32, BATCH), dim3(32, 8, 1), 0, stream>>>(
        vb, vT, SEQ, DMODEL);

    // scores = (q . k^T) / sqrt(D)   (k rows ARE the B-fragments: no transpose)
    gemm(qb, (long long)SEQ * DMODEL, kb, (long long)SEQ * DMODEL,
         nullptr, 0.04419417382415922f, 0, 0,
         sc, nullptr, (long long)SEQ * SEQ, SEQ, SEQ, DMODEL, BATCH);

    // masked softmax -> bf16 attn
    k_softmax<<<dim3(BATCH * SEQ), dim3(256), 0, stream>>>(sc, at);

    // o_in = attn @ v
    gemm(at, (long long)SEQ * SEQ, vT, (long long)DMODEL * SEQ,
         nullptr, 1.0f, 0, 0,
         nullptr, oin, (long long)SEQ * DMODEL, SEQ, DMODEL, SEQ, BATCH);

    // o = o_in @ Wo + bo  (fp32 for residual)
    gemm(oin, 0, WoT, 0, bo, 1.0f, 0, 0, of, nullptr, 0, MTOT, DMODEL, DMODEL, 1);

    // h1 = LN(h0 + o)
    k_add_ln<<<dim3(MTOT), dim3(128), 0, stream>>>(h0f, of, ln1g, ln1b, h1f, h1b);

    // FFN
    gemm(h1b, 0, W1T, 0, b1, 1.0f, 1, 0, nullptr, fm, 0, MTOT, DFF, DMODEL, 1);
    gemm(fm, 0, W2T, 0, b2, 1.0f, 0, 0, ff, nullptr, 0, MTOT, DMODEL, DFF, 1);

    // h2 = LN(h1 + f)  (only bf16 needed for the head GEMM)
    k_add_ln<<<dim3(MTOT), dim3(128), 0, stream>>>(h1f, ff, ln2g, ln2b, nullptr, h2b);

    // logits = h2 @ Wh + bh  -> d_out (fp32, 512 MB, write-once: NT stores)
    gemm(h2b, 0, WhT, 0, bh, 1.0f, 0, 1, out, nullptr, 0, MTOT, VOCAB, DMODEL, 1);
}
